// MAryStem_84911503442396
// MI455X (gfx1250) — compile-verified, matrix operand order checked by reading, not run on hardware
//
#include <hip/hip_runtime.h>
#include <math.h>

// ---------------------------------------------------------------------------
// MI455X / gfx1250 implementation of the BiLSTM fence/segment model.
//   B=64, S=512, D=512, H=256 (per direction), NS=S+1=513, gates G=4H=1024
//
// Design (compile-only; reasoning from MI455X specs):
//  * All big GEMMs (X@Wih^T, l1, dom, sub) -> bf16 WMMA (v_wmma_f32_16x16x32_bf16)
//    with f32 accumulate; tiles staged into LDS with CDNA5 async global->LDS
//    copies (global_load_async_to_lds_b128, ASYNCcnt) in a double-buffered
//    pipeline so DMA copy of tile k+1 overlaps the WMMAs of tile k.
//  * LSTM recurrence: 2 persistent workgroups (one per direction), 1024 threads
//    (32 wave32's). CDNA5's 320KB/WGP LDS holds the full per-step gate matrix
//    [64x1024] f32 (257KB padded) + bf16 recurrent h [64x264] (33KB) -> zero
//    global traffic for state; Whh (512KB bf16) streams from L2 each step.
//  * Tail (scan/condense/blocky-softmax/pool) is < 1GB of traffic vs 23.3TB/s,
//    plain VALU kernels.
// ---------------------------------------------------------------------------

#define B_    64
#define S_    512
#define D_    512
#define H_    256
#define NS_   513
#define G_    1024
#define BSN   (B_ * NS_)        // 32832
#define MP_L1 32896             // BSN padded to multiple of 128
#define BS    (B_ * S_)         // 32768
#define FMAXV 3.4028234663852886e38f
#define FMINV (-3.4028234663852886e38f)

typedef __attribute__((ext_vector_type(16))) __bf16          bf16x16;
typedef __attribute__((ext_vector_type(8)))  float           f32x8;
typedef __attribute__((ext_vector_type(8)))  unsigned short  u16x8;

// --- CDNA5 async global->LDS copy (ISA §15.18.3 op 98), 16B per lane -------
// vdst VGPR holds the LDS byte address; generic LDS pointers carry the LDS
// offset in their low 32 bits.
__device__ __forceinline__ void async_copy16(void* lds_dst, const void* gsrc) {
  asm volatile("global_load_async_to_lds_b128 %0, %1, off"
               :: "v"((unsigned)(size_t)lds_dst),
                  "v"((unsigned long long)(size_t)gsrc)
               : "memory");
}
#define WAIT_ASYNC0() asm volatile("s_wait_asynccnt 0x0" ::: "memory")

// --- fragment load matching ISA 16-bit A/B layout -------------------------
// lanes 0-15: K = {0..7, 16..23}; lanes 16-31: K = {8..15, 24..31}
// B-matrix (K x N) fragment == A-layout applied to W[N,K] rows (lane = n).
__device__ __forceinline__ bf16x16 load_frag(const __bf16* __restrict__ base,
                                             int stride, int row, int k0) {
  const int lane = threadIdx.x & 31;
  const __bf16* p = base + (size_t)(row + (lane & 15)) * stride
                         + k0 + ((lane >> 4) << 3);
  bf16x16 f;
  ((u16x8*)&f)[0] = *(const u16x8*)(p);
  ((u16x8*)&f)[1] = *(const u16x8*)(p + 16);
  return f;
}

__device__ __forceinline__ float sigf(float x) {
  return 1.0f / (1.0f + __expf(-x));
}

__device__ __forceinline__ void atomicMaxF(float* addr, float val) {
  int* ai = (int*)addr;
  int old = __float_as_int(*addr);
  while (__int_as_float(old) < val) {
    int assumed = old;
    old = atomicCAS(ai, assumed, __float_as_int(val));
    if (old == assumed) break;
  }
}

// ---------------------------------------------------------------------------
// Generic tiled WMMA GEMM:  C[M,N] = act(A[M,K] @ W[N,K]^T + bias[N])
// Workgroup = 256 thr (8 waves), tile 128(M) x 64(N), K-step 32.
// Double-buffered LDS; tiles staged by async global->LDS DMA so the copy of
// tile k+1 overlaps the WMMAs of tile k (ASYNCcnt-gated).
// M must be a multiple of 128, N of 64, K of 32 (all call sites padded).
// ---------------------------------------------------------------------------
__global__ __launch_bounds__(256) void k_gemm(const __bf16* __restrict__ A,
                                              const __bf16* __restrict__ Bw,
                                              float* __restrict__ C,
                                              const float* __restrict__ bias,
                                              int M, int N, int K, int act) {
  __shared__ __bf16 As[2][128][40];   // stride 40 el = 80B (16B-aligned rows)
  __shared__ __bf16 Bs[2][64][40];
  const int tid = threadIdx.x;
  const int wid = tid >> 5;
  const int m0  = blockIdx.y * 128;
  const int n0  = blockIdx.x * 64;
  const int wm  = (wid >> 1) * 32;
  const int wn  = (wid & 1) * 32;

  const int sr = tid >> 1;           // staging row
  const int sc = (tid & 1) << 4;     // staging col chunk

  f32x8 acc[2][2] = {};

  // ---- prologue: async-stage tile 0 into buffer 0 ----
  {
    const __bf16* gA = A + (size_t)(m0 + sr) * K + sc;
    async_copy16(&As[0][sr][sc],     gA);
    async_copy16(&As[0][sr][sc + 8], gA + 8);
    if (tid < 128) {
      const __bf16* gB = Bw + (size_t)(n0 + sr) * K + sc;
      async_copy16(&Bs[0][sr][sc],     gB);
      async_copy16(&Bs[0][sr][sc + 8], gB + 8);
    }
  }
  WAIT_ASYNC0();
  __syncthreads();

  int cur = 0;
  for (int k0 = 0; k0 < K; k0 += 32) {
    const int nxt = cur ^ 1;
    const bool more = (k0 + 32 < K);

    // ---- issue async DMA for tile k+1 (overlaps WMMAs below) ----
    if (more) {
      const __bf16* gA = A + (size_t)(m0 + sr) * K + (k0 + 32) + sc;
      async_copy16(&As[nxt][sr][sc],     gA);
      async_copy16(&As[nxt][sr][sc + 8], gA + 8);
      if (k0 + 64 < K)  // -> global_prefetch_b8, pull tile k+2 toward L2
        __builtin_prefetch((const void*)(gA + 32), 0, 1);
      if (tid < 128) {
        const __bf16* gB = Bw + (size_t)(n0 + sr) * K + (k0 + 32) + sc;
        async_copy16(&Bs[nxt][sr][sc],     gB);
        async_copy16(&Bs[nxt][sr][sc + 8], gB + 8);
      }
    }

    // ---- compute from buffer `cur` ----
    bf16x16 a0 = load_frag(&As[cur][0][0], 40, wm,      0);
    bf16x16 a1 = load_frag(&As[cur][0][0], 40, wm + 16, 0);
    bf16x16 b0 = load_frag(&Bs[cur][0][0], 40, wn,      0);
    bf16x16 b1 = load_frag(&Bs[cur][0][0], 40, wn + 16, 0);
    acc[0][0] = __builtin_amdgcn_wmma_f32_16x16x32_bf16(false, a0, false, b0, (short)0, acc[0][0], false, false);
    acc[0][1] = __builtin_amdgcn_wmma_f32_16x16x32_bf16(false, a0, false, b1, (short)0, acc[0][1], false, false);
    acc[1][0] = __builtin_amdgcn_wmma_f32_16x16x32_bf16(false, a1, false, b0, (short)0, acc[1][0], false, false);
    acc[1][1] = __builtin_amdgcn_wmma_f32_16x16x32_bf16(false, a1, false, b1, (short)0, acc[1][1], false, false);

    // ---- retire this iteration's DMA, publish buffer `nxt` ----
    if (more) WAIT_ASYNC0();
    __syncthreads();
    cur = nxt;
  }

  // epilogue: C layout per ISA (VGPR r -> M=r / r+8, lane -> N)
  const int lane  = tid & 31;
  const int rbase = (lane >> 4) << 3;
  const int cN    = lane & 15;
#pragma unroll
  for (int i = 0; i < 2; ++i)
#pragma unroll
    for (int j = 0; j < 2; ++j) {
      int col = n0 + wn + j * 16 + cN;
      float bv = bias ? bias[col] : 0.0f;
#pragma unroll
      for (int r = 0; r < 8; ++r) {
        int row = m0 + wm + i * 16 + rbase + r;
        float v = acc[i][j][r] + bv;
        if (act == 1) v = tanhf(v);
        C[(size_t)row * N + col] = v;
      }
    }
}

// ---------------------------------------------------------------------------
// Persistent LSTM recurrence, one workgroup per direction (blockIdx.x = dir).
// 1024 threads = 32 waves. Per step: gates[64,1024] = h[64,256] @ Whh^T (+gx),
// WMMA from LDS-resident bf16 h; gating VALU updates c (registers) and h.
// LDS: hbf [64][264] bf16 (33,792B) + gl [64][1028] f32 (263,168B) = 296,960B
//      -> exploits CDNA5's 320KB-per-workgroup LDS.
// ---------------------------------------------------------------------------
__global__ __launch_bounds__(1024) void k_lstm(const float* __restrict__ gx_f,
                                               const float* __restrict__ gx_b,
                                               const __bf16* __restrict__ Whh_f,
                                               const __bf16* __restrict__ Whh_b,
                                               const float* __restrict__ h0,
                                               const float* __restrict__ c0,
                                               float* __restrict__ fw_h,
                                               float* __restrict__ bw_h) {
  const int dir = blockIdx.x;
  const float*  gx   = dir ? gx_b   : gx_f;
  const __bf16* Whh  = dir ? Whh_b  : Whh_f;
  float*        hout = dir ? bw_h   : fw_h;

  extern __shared__ char smem[];
  __bf16* hbf = (__bf16*)smem;                                 // [64][264]
  float*  gl  = (float*)(smem + 64 * 264 * sizeof(__bf16));    // [64][1028]

  const int tid  = threadIdx.x;
  const int wid  = tid >> 5;
  const int lane = tid & 31;

  // per-thread cell state: batch tb, hidden cols j0 + 16u
  const int tb = tid >> 4;
  const int j0 = tid & 15;
  float creg[16];
#pragma unroll
  for (int u = 0; u < 16; ++u) {
    int j = j0 + (u << 4);
    creg[u] = c0[dir * H_ + j];
    hbf[tb * 264 + j] = (__bf16)tanhf(h0[dir * H_ + j]);
  }
  __syncthreads();

  const int wm = (wid >> 3) << 4;    // M tile base (0,16,32,48)
  const int nb = (wid & 7) * 128;    // 8 consecutive N tiles per wave

  for (int t = 0; t < S_; ++t) {
    const int trow = dir ? (S_ - 1 - t) : t;

    f32x8 acc[8] = {};
    for (int k0 = 0; k0 < H_; k0 += 32) {
      bf16x16 a = load_frag(hbf, 264, wm, k0);
#pragma unroll
      for (int i = 0; i < 8; ++i) {
        bf16x16 b = load_frag(Whh, H_, nb + i * 16, k0);  // hits L2/WGP$
        acc[i] = __builtin_amdgcn_wmma_f32_16x16x32_bf16(false, a, false, b, (short)0, acc[i], false, false);
      }
    }

    // spill gates (+ precomputed input projection) to LDS
    const int rbase = (lane >> 4) << 3;
    const int cN    = lane & 15;
#pragma unroll
    for (int i = 0; i < 8; ++i) {
      int col = nb + i * 16 + cN;
#pragma unroll
      for (int r = 0; r < 8; ++r) {
        int brow = wm + rbase + r;
        gl[brow * 1028 + col] =
            acc[i][r] + gx[((size_t)brow * S_ + trow) * G_ + col];
      }
    }
    __syncthreads();

    // gating (PyTorch order i,f,g,o)
#pragma unroll
    for (int u = 0; u < 16; ++u) {
      int j = j0 + (u << 4);
      float ig = gl[tb * 1028 + j];
      float fg = gl[tb * 1028 + 256 + j];
      float gg = gl[tb * 1028 + 512 + j];
      float og = gl[tb * 1028 + 768 + j];
      float c  = sigf(fg) * creg[u] + sigf(ig) * tanhf(gg);
      float h  = sigf(og) * tanhf(c);
      creg[u] = c;
      hbf[tb * 264 + j] = (__bf16)h;
      hout[((size_t)tb * S_ + trow) * H_ + j] = h;
    }
    __syncthreads();
  }
}

// --------------------------- small support kernels -------------------------
__global__ void k_zero(float* __restrict__ p, size_t n) {
  size_t i = blockIdx.x * (size_t)blockDim.x + threadIdx.x;
  if (i < n) p[i] = 0.0f;
}

__global__ void k_cvt(const float* __restrict__ s, __bf16* __restrict__ d, size_t n) {
  size_t i = blockIdx.x * (size_t)blockDim.x + threadIdx.x;
  if (i < n) d[i] = (__bf16)s[i];
}

__global__ void k_bias2(const float* __restrict__ a, const float* __restrict__ b,
                        float* __restrict__ o, int n) {
  int i = blockIdx.x * blockDim.x + threadIdx.x;
  if (i < n) o[i] = a[i] + b[i];
}

__global__ void k_wcomb(const float* __restrict__ sfwd, const float* __restrict__ sbwd,
                        const float* __restrict__ sfw,  const float* __restrict__ sbw,
                        __bf16* __restrict__ W) {
  int idx = blockIdx.x * blockDim.x + threadIdx.x;   // D_*1024
  if (idx >= D_ * 1024) return;
  int n = idx >> 10, k = idx & 1023;
  float v;
  if      (k < 256) v = sfwd[n * 256 + k];
  else if (k < 512) v = sbwd[n * 256 + (k - 256)];
  else if (k < 768) v = sfw [n * 256 + (k - 512)];
  else              v = sbw [n * 256 + (k - 768)];
  W[idx] = (__bf16)v;
}

__global__ void k_seqlen(const unsigned char* __restrict__ ex, int* __restrict__ sl) {
  int b = blockIdx.x, tid = threadIdx.x;
  __shared__ int cnt;
  if (tid == 0) cnt = 0;
  __syncthreads();
  int loc = 0;
  for (int s = tid; s < S_; s += 256) loc += ex[b * S_ + s] ? 1 : 0;
  atomicAdd(&cnt, loc);
  __syncthreads();
  if (tid == 0) sl[b] = cnt;
}

// build fencepost arrays fw/bw [B,NS,H] and bf16 A for the l1 GEMM (padded M)
__global__ void k_fposts(const float* __restrict__ fw_h, const float* __restrict__ bw_h,
                         const float* __restrict__ pad,  const int* __restrict__ sl,
                         float* __restrict__ fw, float* __restrict__ bw,
                         __bf16* __restrict__ A_l1) {
  const int row = blockIdx.x;   // 0..MP_L1-1
  const int j   = threadIdx.x;  // 0..255
  if (row >= BSN) {
    A_l1[(size_t)row * 512 + j]       = (__bf16)0.0f;
    A_l1[(size_t)row * 512 + 256 + j] = (__bf16)0.0f;
    return;
  }
  const int b = row / NS_, p = row % NS_;
  const float padt = tanhf(pad[j]);
  float fv = (p == 0) ? padt : fw_h[((size_t)b * S_ + (p - 1)) * H_ + j];
  float bv = (p == S_ || p == sl[b]) ? padt
                                     : bw_h[((size_t)b * S_ + p) * H_ + j];
  fw[((size_t)b * NS_ + p) * H_ + j] = fv;
  bw[((size_t)b * NS_ + p) * H_ + j] = bv;
  A_l1[(size_t)row * 512 + j]       = (__bf16)fv;
  A_l1[(size_t)row * 512 + 256 + j] = (__bf16)bv;
}

// fence logits = h1 @ W_l2^T + b_l2 (one wave32 per row, shuffle reduction)
__global__ void k_l2dot(const float* __restrict__ h1, const float* __restrict__ W2,
                        const float* __restrict__ b2, float* __restrict__ logits) {
  int wid = threadIdx.x >> 5, lane = threadIdx.x & 31;
  int row = blockIdx.x * 8 + wid;
  if (row >= BSN) return;
  float p = 0.0f;
  for (int i = lane; i < H_; i += 32) p += h1[(size_t)row * H_ + i] * W2[i];
#pragma unroll
  for (int off = 16; off > 0; off >>= 1) p += __shfl_xor(p, off, 32);
  if (lane == 0) logits[row] = p + b2[0];
}

__global__ void k_thresh(float* __restrict__ logits, const int* __restrict__ sl,
                         float* __restrict__ out_fl) {
  int idx = blockIdx.x * blockDim.x + threadIdx.x;
  if (idx >= BSN) return;
  int b = idx / NS_, p = idx % NS_;
  float v = logits[idx];
  int s = sl[b];
  if (p == 0 || p == s) v = FMAXV;
  if (p > s)            v = FMINV;
  logits[idx] = v;
  out_fl[idx] = v;
}

// left-pack fence indices + 1-based section ids (serial per batch row)
__global__ void k_condense(const float* __restrict__ logits, const int* __restrict__ sl,
                           int* __restrict__ fidx, int* __restrict__ sections,
                           float* __restrict__ out_fidx, float* __restrict__ out_fbool) {
  int b = threadIdx.x;
  if (b >= B_) return;
  int s = sl[b];
  int cnt = 0, cum = 0;
  for (int p = 0; p < NS_; ++p) {
    bool f = logits[b * NS_ + p] > 0.0f;
    if (f) fidx[b * NS_ + cnt++] = p;
    cum += f ? 1 : 0;
    if (p < S_) sections[b * S_ + p] = (p < s) ? cum : 0;
  }
  for (int k = cnt; k < NS_; ++k) fidx[b * NS_ + k] = 0;
  for (int p = 0; p < NS_; ++p) out_fidx[b * NS_ + p] = (float)fidx[b * NS_ + p];
  for (int t = 0; t < S_; ++t)
    out_fbool[b * S_ + t] = (fidx[b * NS_ + 1 + t] > 0) ? 1.0f : 0.0f;
}

// gather fencepost spans -> bf16 GEMM inputs for dom (K=512) and sub (K=1024)
__global__ void k_dsin(const float* __restrict__ fw, const float* __restrict__ bw,
                       const int* __restrict__ fidx,
                       __bf16* __restrict__ dom_in, __bf16* __restrict__ sub_in) {
  int token = blockIdx.x;       // b*S+s
  int j = threadIdx.x;          // 0..255
  int b = token >> 9, s = token & (S_ - 1);
  int f0 = fidx[b * NS_ + s], f1 = fidx[b * NS_ + s + 1];
  const float* fwb = fw + (size_t)b * NS_ * H_;
  const float* bwb = bw + (size_t)b * NS_ * H_;
  float fg = fwb[(size_t)f1 * H_ + j] - fwb[(size_t)f0 * H_ + j];
  float bg = bwb[(size_t)f0 * H_ + j] - bwb[(size_t)f1 * H_ + j];
  dom_in[(size_t)token * 512 + j]       = (__bf16)fg;
  dom_in[(size_t)token * 512 + 256 + j] = (__bf16)bg;
  float fprev = fwb[(size_t)s * H_ + j];
  float fnext = fwb[(size_t)(s + 1) * H_ + j];
  float bprev = bwb[(size_t)s * H_ + j];
  float bnext = bwb[(size_t)(s + 1) * H_ + j];
  sub_in[(size_t)token * 1024 + j]       = (__bf16)(fnext - fprev);
  sub_in[(size_t)token * 1024 + 256 + j] = (__bf16)(bprev - bnext);
  sub_in[(size_t)token * 1024 + 512 + j] = (__bf16)fprev;
  sub_in[(size_t)token * 1024 + 768 + j] = (__bf16)bnext;
}

// per-token attention logit = <sub_emb[token], dom_emb[b, sec-1]>
__global__ void k_toklogit(const float* __restrict__ sub_e, const float* __restrict__ dom_e,
                           const int* __restrict__ sections, float* __restrict__ tlog) {
  int wid = threadIdx.x >> 5, lane = threadIdx.x & 31;
  int token = blockIdx.x * 8 + wid;
  if (token >= BS) return;
  int b = token >> 9;
  int sec = sections[token];
  if (sec <= 0) { if (lane == 0) tlog[token] = FMINV; return; }
  const float* dt = dom_e + ((size_t)(b * S_ + sec - 1)) * D_;
  const float* se = sub_e + (size_t)token * D_;
  float p = 0.0f;
  for (int i = lane; i < D_; i += 32) p += se[i] * dt[i];
#pragma unroll
  for (int off = 16; off > 0; off >>= 1) p += __shfl_xor(p, off, 32);
  if (lane == 0) tlog[token] = p;
}

// per-row blocky softmax (segment max / sum in LDS)
__global__ void k_blocksoft(const float* __restrict__ tlog, const int* __restrict__ sections,
                            float* __restrict__ wout, float* __restrict__ wws) {
  __shared__ float smax[NS_ + 1];
  __shared__ float ssum[NS_ + 1];
  int b = blockIdx.x, tid = threadIdx.x;
  for (int i = tid; i <= NS_; i += 256) { smax[i] = FMINV; ssum[i] = 0.0f; }
  __syncthreads();
  for (int s = tid; s < S_; s += 256) {
    int sec = sections[b * S_ + s];
    if (sec > 0) atomicMaxF(&smax[sec], tlog[b * S_ + s]);
  }
  __syncthreads();
  float evals[2];
#pragma unroll
  for (int r = 0; r < 2; ++r) {
    int s = tid + r * 256;
    int sec = sections[b * S_ + s];
    float e = 0.0f;
    if (sec > 0) { e = __expf(tlog[b * S_ + s] - smax[sec]); atomicAdd(&ssum[sec], e); }
    evals[r] = e;
  }
  __syncthreads();
#pragma unroll
  for (int r = 0; r < 2; ++r) {
    int s = tid + r * 256;
    int sec = sections[b * S_ + s];
    float w = (sec > 0) ? evals[r] / fmaxf(ssum[sec], 1e-9f) : 0.0f;
    wout[b * S_ + s] = w;
    wws [b * S_ + s] = w;
  }
}

// pooled[b, sec-1, :] += w * unit_emb[b, s, :]
__global__ void k_pool(const float* __restrict__ ue, const float* __restrict__ w,
                       const int* __restrict__ sections, float* __restrict__ pooled) {
  int token = blockIdx.x, tid = threadIdx.x;
  int b = token >> 9;
  int sec = sections[token];
  if (sec <= 0) return;
  float wv = w[token];
  if (wv == 0.0f) return;
  size_t orow = ((size_t)(b * S_ + (sec - 1))) * D_;
  size_t irow = (size_t)token * D_;
  atomicAdd(&pooled[orow + tid],       wv * ue[irow + tid]);
  atomicAdd(&pooled[orow + tid + 256], wv * ue[irow + tid + 256]);
}

// ---------------------------------------------------------------------------
extern "C" void kernel_launch(void* const* d_in, const int* in_sizes, int n_in,
                              void* d_out, int out_size, void* d_ws, size_t ws_size,
                              hipStream_t stream) {
  (void)in_sizes; (void)n_in; (void)out_size; (void)ws_size;

  const float* ue    = (const float*)d_in[0];
  const unsigned char* ex = (const unsigned char*)d_in[1];
  const float* Wih_f = (const float*)d_in[2];
  const float* Whh_f = (const float*)d_in[3];
  const float* bih_f = (const float*)d_in[4];
  const float* bhh_f = (const float*)d_in[5];
  const float* Wih_b = (const float*)d_in[6];
  const float* Whh_b = (const float*)d_in[7];
  const float* bih_b = (const float*)d_in[8];
  const float* bhh_b = (const float*)d_in[9];
  const float* h0    = (const float*)d_in[10];
  const float* c0    = (const float*)d_in[11];
  const float* pad   = (const float*)d_in[12];
  const float* W_l1  = (const float*)d_in[13];
  const float* b_l1  = (const float*)d_in[14];
  const float* W_l2  = (const float*)d_in[15];
  const float* b_l2  = (const float*)d_in[16];
  const float* W_dom = (const float*)d_in[17];
  const float* b_dom = (const float*)d_in[18];
  const float* W_sfw = (const float*)d_in[19];
  const float* b_sfw = (const float*)d_in[20];
  const float* W_sbw = (const float*)d_in[21];
  const float* W_sfwd= (const float*)d_in[22];
  const float* W_sbwd= (const float*)d_in[23];

  // output layout (flat, return order)
  float* o_flog  = (float*)d_out;                 // [B,NS]
  float* o_w     = o_flog + BSN;                  // [B,S]
  float* o_pool  = o_w + BS;                      // [B,S,D]
  float* o_fidx  = o_pool + (size_t)BS * D_;      // [B,NS]
  float* o_fbool = o_fidx + BSN;                  // [B,S]

  // workspace carve-up
  size_t off = 0;
  auto alloc = [&](size_t bytes) -> void* {
    void* p = (char*)d_ws + off;
    off += (bytes + 255) & ~(size_t)255;
    return p;
  };
  __bf16* ue_bf    = (__bf16*)alloc((size_t)BS * D_ * 2);
  __bf16* wihf_bf  = (__bf16*)alloc((size_t)G_ * D_ * 2);
  __bf16* wihb_bf  = (__bf16*)alloc((size_t)G_ * D_ * 2);
  __bf16* whhf_bf  = (__bf16*)alloc((size_t)G_ * H_ * 2);
  __bf16* whhb_bf  = (__bf16*)alloc((size_t)G_ * H_ * 2);
  float*  biasf    = (float*)alloc(G_ * 4);
  float*  biasb    = (float*)alloc(G_ * 4);
  float*  gx_f     = (float*)alloc((size_t)BS * G_ * 4);
  float*  gx_b     = (float*)alloc((size_t)BS * G_ * 4);
  float*  fw_h     = (float*)alloc((size_t)BS * H_ * 4);
  float*  bw_h     = (float*)alloc((size_t)BS * H_ * 4);
  float*  fwp      = (float*)alloc((size_t)BSN * H_ * 4);
  float*  bwp      = (float*)alloc((size_t)BSN * H_ * 4);
  __bf16* A_l1     = (__bf16*)alloc((size_t)MP_L1 * 512 * 2);
  __bf16* wl1_bf   = (__bf16*)alloc((size_t)H_ * 512 * 2);
  float*  h1       = (float*)alloc((size_t)MP_L1 * H_ * 4);
  float*  flog     = (float*)alloc((size_t)BSN * 4);
  int*    fidx     = (int*)alloc((size_t)BSN * 4);
  int*    sections = (int*)alloc((size_t)BS * 4);
  int*    seqlen   = (int*)alloc(B_ * 4);
  __bf16* dom_in   = (__bf16*)alloc((size_t)BS * 512 * 2);
  __bf16* wdom_bf  = (__bf16*)alloc((size_t)D_ * 512 * 2);
  __bf16* sub_in   = (__bf16*)alloc((size_t)BS * 1024 * 2);
  __bf16* wcomb_bf = (__bf16*)alloc((size_t)D_ * 1024 * 2);
  float*  dom_e    = (float*)alloc((size_t)BS * D_ * 4);
  float*  sub_e    = (float*)alloc((size_t)BS * D_ * 4);
  float*  tlog     = (float*)alloc((size_t)BS * 4);
  float*  wws      = (float*)alloc((size_t)BS * 4);

  // ---- housekeeping (graph-replay determinism: pooled is accumulated) ----
  k_zero<<<(int)(((size_t)BS * D_ + 255) / 256), 256, 0, stream>>>(o_pool, (size_t)BS * D_);

  // ---- precision conversion (f32 -> bf16 for WMMA operands) ----
  k_cvt<<<(int)(((size_t)BS * D_ + 255) / 256), 256, 0, stream>>>(ue, ue_bf, (size_t)BS * D_);
  k_cvt<<<(G_ * D_ + 255) / 256, 256, 0, stream>>>(Wih_f, wihf_bf, (size_t)G_ * D_);
  k_cvt<<<(G_ * D_ + 255) / 256, 256, 0, stream>>>(Wih_b, wihb_bf, (size_t)G_ * D_);
  k_cvt<<<(G_ * H_ + 255) / 256, 256, 0, stream>>>(Whh_f, whhf_bf, (size_t)G_ * H_);
  k_cvt<<<(G_ * H_ + 255) / 256, 256, 0, stream>>>(Whh_b, whhb_bf, (size_t)G_ * H_);
  k_cvt<<<(H_ * 512 + 255) / 256, 256, 0, stream>>>(W_l1, wl1_bf, (size_t)H_ * 512);
  k_cvt<<<(D_ * 512 + 255) / 256, 256, 0, stream>>>(W_dom, wdom_bf, (size_t)D_ * 512);
  k_wcomb<<<(D_ * 1024 + 255) / 256, 256, 0, stream>>>(W_sfwd, W_sbwd, W_sfw, W_sbw, wcomb_bf);
  k_bias2<<<(G_ + 255) / 256, 256, 0, stream>>>(bih_f, bhh_f, biasf, G_);
  k_bias2<<<(G_ + 255) / 256, 256, 0, stream>>>(bih_b, bhh_b, biasb, G_);
  k_seqlen<<<B_, 256, 0, stream>>>(ex, seqlen);

  // ---- time-parallel input projections: gx = X @ Wih^T + (bih+bhh) ----
  k_gemm<<<dim3(G_ / 64, BS / 128), 256, 0, stream>>>(ue_bf, wihf_bf, gx_f, biasf, BS, G_, D_, 0);
  k_gemm<<<dim3(G_ / 64, BS / 128), 256, 0, stream>>>(ue_bf, wihb_bf, gx_b, biasb, BS, G_, D_, 0);

  // ---- persistent bidirectional recurrence (LDS-resident state) ----
  const int lstm_lds = 64 * 264 * 2 + 64 * 1028 * 4;   // 296,960 B < 320 KB WGP LDS
  k_lstm<<<2, 1024, lstm_lds, stream>>>(gx_f, gx_b, whhf_bf, whhb_bf, h0, c0, fw_h, bw_h);

  // ---- fenceposts + l1 MLP + fence logits ----
  k_fposts<<<MP_L1, 256, 0, stream>>>(fw_h, bw_h, pad, seqlen, fwp, bwp, A_l1);
  k_gemm<<<dim3(H_ / 64, MP_L1 / 128), 256, 0, stream>>>(A_l1, wl1_bf, h1, b_l1, MP_L1, H_, 512, 1);
  k_l2dot<<<(BSN + 7) / 8, 256, 0, stream>>>(h1, W_l2, b_l2, flog);
  k_thresh<<<(BSN + 255) / 256, 256, 0, stream>>>(flog, seqlen, o_flog);
  k_condense<<<1, 64, 0, stream>>>(flog, seqlen, fidx, sections, o_fidx, o_fbool);

  // ---- span gathers + dom/sub projections ----
  k_dsin<<<BS, 256, 0, stream>>>(fwp, bwp, fidx, dom_in, sub_in);
  k_gemm<<<dim3(D_ / 64, BS / 128), 256, 0, stream>>>(dom_in, wdom_bf, dom_e, b_dom, BS, D_, 512, 0);
  k_gemm<<<dim3(D_ / 64, BS / 128), 256, 0, stream>>>(sub_in, wcomb_bf, sub_e, b_sfw, BS, D_, 1024, 0);

  // ---- blocky softmax + pooled ----
  k_toklogit<<<(BS + 7) / 8, 256, 0, stream>>>(sub_e, dom_e, sections, tlog);
  k_blocksoft<<<B_, 256, 0, stream>>>(tlog, sections, o_w, wws);
  k_pool<<<BS, 256, 0, stream>>>(ue, wws, sections, o_pool);
}
